// EGESModel_83150566850865
// MI455X (gfx1250) — matrix operand order repaired and checked by reading, not compiled, not run on hardware
//
#include <hip/hip_runtime.h>
#include <hip/hip_bf16.h>
#include <math.h>

// CDNA5 (gfx1250) EGES kernel: gather-bound embedding combine + WMMA scoring.
// One wave32 handles a tile of 16 batch elements. Scoring logits[b,c] for the
// tile is computed as diag(H x Ctx_c^T) via 4 chained v_wmma_f32_16x16x32_bf16
// per context slot (f32 accumulate), then sigmoid on the diagonal.

typedef __attribute__((ext_vector_type(16))) __bf16 v16bf;
typedef __attribute__((ext_vector_type(8)))  float  v8f;

#define DIM 128
#define NCTX 6

static __device__ __forceinline__ __bf16 f2bf(float f) {
    return (__bf16)f;  // v_cvt to bf16 (RNE)
}

__global__ __launch_bounds__(256)
void eges_wmma_kernel(const int* __restrict__ central,
                      const int* __restrict__ side0,
                      const int* __restrict__ side1,     // [B,5]
                      const int* __restrict__ ctxidx,    // [B,6]
                      const float* __restrict__ Win,     // [NUM_ITEMS,128]
                      const float* __restrict__ Wout,    // [NUM_ITEMS,128]
                      const float* __restrict__ Ww,      // [NUM_ITEMS,3]
                      const float* __restrict__ Ws0,     // [V0,128]
                      const float* __restrict__ Ws1,     // [V1,128]
                      float* __restrict__ out,           // [B,6]
                      int B)
{
    const int lane = threadIdx.x & 31;
    const int wave = threadIdx.x >> 5;
    const int tile = blockIdx.x * (blockDim.x >> 5) + wave;
    const int b0   = tile * 16;
    if (b0 >= B) return;                       // wave-uniform exit

    const int mrow = lane & 15;                // batch row this lane serves
    const int hi   = lane >> 4;                // 0: lanes 0-15, 1: lanes 16-31
    int b = b0 + mrow; if (b >= B) b = B - 1;  // clamp (keeps EXEC all-ones for WMMA)

    // ---- per-batch indices (lane m and m+16 duplicate; L0 hits) ----
    const int ci  = central[b];
    const int s0i = side0[b];
    int s1i[5];
#pragma unroll
    for (int j = 0; j < 5; ++j) s1i[j] = side1[b * 5 + j];
    const float* ctr[NCTX];
#pragma unroll
    for (int c = 0; c < NCTX; ++c)
        ctr[c] = Wout + (size_t)ctxidx[b * NCTX + c] * DIM;

    // ---- softmax over the 3 combine weights ----
    const float* wp = Ww + (size_t)ci * 3;
    const float x0 = wp[0], x1 = wp[1], x2 = wp[2];
    const float mx = fmaxf(x0, fmaxf(x1, x2));
    const float e0 = expf(x0 - mx), e1 = expf(x1 - mx), e2 = expf(x2 - mx);
    const float inv = 1.0f / (e0 + e1 + e2);
    const float w0 = e0 * inv, w1 = e1 * inv, w2m = e2 * inv * 0.2f; // mean of 5 folded in

    const float* itr = Win + (size_t)ci  * DIM;
    const float* s0r = Ws0 + (size_t)s0i * DIM;
    const float* s1r[5];
#pragma unroll
    for (int j = 0; j < 5; ++j) s1r[j] = Ws1 + (size_t)s1i[j] * DIM;

    v8f acc[NCTX] = {};   // f32 16x16 accumulators, one per context slot

#pragma unroll
    for (int i = 0; i < 4; ++i) {              // K chunks of 32 along D
        // ---- build A fragment: hidden rows, 16-bit A 16x32 layout ----
        // lane<16: K = {0..7, 16..23}+32i ; lane>=16: K = {8..15, 24..31}+32i
        const int baseA = 32 * i + hi * 8;
        float h[16];
#pragma unroll
        for (int g = 0; g < 2; ++g) {
            const int off = baseA + g * 16;
            const float4 it0 = *(const float4*)(itr + off);
            const float4 it1 = *(const float4*)(itr + off + 4);
            const float4 sa0 = *(const float4*)(s0r + off);
            const float4 sa1 = *(const float4*)(s0r + off + 4);
            float4 u0 = make_float4(0.f, 0.f, 0.f, 0.f);
            float4 u1 = make_float4(0.f, 0.f, 0.f, 0.f);
#pragma unroll
            for (int j = 0; j < 5; ++j) {
                const float4 t0 = *(const float4*)(s1r[j] + off);
                const float4 t1 = *(const float4*)(s1r[j] + off + 4);
                u0.x += t0.x; u0.y += t0.y; u0.z += t0.z; u0.w += t0.w;
                u1.x += t1.x; u1.y += t1.y; u1.z += t1.z; u1.w += t1.w;
            }
            h[g * 8 + 0] = w0 * it0.x + w1 * sa0.x + w2m * u0.x;
            h[g * 8 + 1] = w0 * it0.y + w1 * sa0.y + w2m * u0.y;
            h[g * 8 + 2] = w0 * it0.z + w1 * sa0.z + w2m * u0.z;
            h[g * 8 + 3] = w0 * it0.w + w1 * sa0.w + w2m * u0.w;
            h[g * 8 + 4] = w0 * it1.x + w1 * sa1.x + w2m * u1.x;
            h[g * 8 + 5] = w0 * it1.y + w1 * sa1.y + w2m * u1.y;
            h[g * 8 + 6] = w0 * it1.z + w1 * sa1.z + w2m * u1.z;
            h[g * 8 + 7] = w0 * it1.w + w1 * sa1.w + w2m * u1.w;
        }
        v16bf a;
#pragma unroll
        for (int k = 0; k < 16; ++k) a[k] = f2bf(h[k]);

        // ---- B fragments: ctx rows, 32x16 B layout ----
        // lane<16: K = {0..7, 16..23}+32i ; lane>=16: K = {8..15, 24..31}+32i
        const int baseB = 32 * i + hi * 8;
#pragma unroll
        for (int c = 0; c < NCTX; ++c) {
            const float4* cr = (const float4*)(ctr[c] + baseB);
            const float4 q0 = cr[0], q1 = cr[1], q2 = cr[4], q3 = cr[5];
            v16bf bv;
            bv[0]  = f2bf(q0.x); bv[1]  = f2bf(q0.y); bv[2]  = f2bf(q0.z); bv[3]  = f2bf(q0.w);
            bv[4]  = f2bf(q1.x); bv[5]  = f2bf(q1.y); bv[6]  = f2bf(q1.z); bv[7]  = f2bf(q1.w);
            bv[8]  = f2bf(q2.x); bv[9]  = f2bf(q2.y); bv[10] = f2bf(q2.z); bv[11] = f2bf(q2.w);
            bv[12] = f2bf(q3.x); bv[13] = f2bf(q3.y); bv[14] = f2bf(q3.z); bv[15] = f2bf(q3.w);
            // D = A x B + C, f32 accumulate
            acc[c] = __builtin_amdgcn_wmma_f32_16x16x32_bf16(
                         false, a, false, bv, (short)0, acc[c], false, false);
        }
    }

    // ---- extract diagonal D[m,m] and emit sigmoid ----
    // lanes 0-7: m = lane, element = lane&7 ; lanes 24-31: m = 8 + (lane&7)
    const int  l8  = lane & 7;
    const bool act = (lane < 8) || (lane >= 24);
    const int  m   = (lane < 8) ? l8 : (8 + l8);
    const int  ob  = b0 + m;
    if (act && ob < B) {
#pragma unroll
        for (int c = 0; c < NCTX; ++c) {
            const float v = acc[c][l8];
            out[ob * NCTX + c] = 1.0f / (1.0f + expf(-v));
        }
    }
}

extern "C" void kernel_launch(void* const* d_in, const int* in_sizes, int n_in,
                              void* d_out, int out_size, void* d_ws, size_t ws_size,
                              hipStream_t stream) {
    const int B = in_sizes[0];                       // 16384
    const int*   central = (const int*)  d_in[0];
    const int*   side0   = (const int*)  d_in[1];
    const int*   side1   = (const int*)  d_in[2];
    const int*   ctxidx  = (const int*)  d_in[3];
    const float* Win     = (const float*)d_in[4];
    const float* Wout    = (const float*)d_in[5];
    const float* Ww      = (const float*)d_in[6];
    const float* Ws0     = (const float*)d_in[7];
    const float* Ws1     = (const float*)d_in[8];
    float* out = (float*)d_out;

    const int waves_per_block = 8;                   // 256 threads, wave32
    const int tiles  = (B + 15) / 16;
    const int blocks = (tiles + waves_per_block - 1) / waves_per_block;
    eges_wmma_kernel<<<blocks, 256, 0, stream>>>(central, side0, side1, ctxidx,
                                                 Win, Wout, Ww, Ws0, Ws1, out, B);
}
